// _HeteroTimingConvNet_72473278153129
// MI455X (gfx1250) — matrix operand order repaired and checked by reading, not compiled, not run on hardware
//
#include <hip/hip_runtime.h>

#define NND   50000
#define NE    600000
#define HID   128
#define INDIM 14
#define NLAY  4
#define NET   3
#define NG    64
#define LN_EPS 1e-5f

typedef __attribute__((ext_vector_type(16))) _Float16 v16h;
typedef __attribute__((ext_vector_type(8)))  float    v8f;
typedef __attribute__((ext_vector_type(4)))  int      v4i;

#if __has_builtin(__builtin_amdgcn_global_load_async_to_lds_b128) && \
    __has_builtin(__builtin_amdgcn_s_wait_asynccnt)
#define USE_ASYNC_LDS 1
typedef __attribute__((address_space(1))) v4i* gas_ptr;   // global int4*
typedef __attribute__((address_space(3))) v4i* las_ptr;   // LDS int4*
#else
#define USE_ASYNC_LDS 0
#endif

// ---------------------------------------------------------------------------
// Generic fill
// ---------------------------------------------------------------------------
__global__ void fill_kernel(float* __restrict__ p, float v, long n) {
    long t = (long)blockIdx.x * blockDim.x + threadIdx.x;
    if (t < n) p[t] = v;
}

// f32 -> f16 weight conversion WITH transpose: out[n*HID+k] = (f16)in[k*HID+n]
// (so the GEMM's LDS image [n][k] is a straight 32KB copy)
__global__ void convert_f16_t_kernel(const float* __restrict__ in,
                                     _Float16* __restrict__ out, int n) {
    int t = blockIdx.x * blockDim.x + threadIdx.x;
    if (t < n) {
        int nn = t >> 7, k = t & 127;
        out[t] = (_Float16)in[k * HID + nn];
    }
}

// ---------------------------------------------------------------------------
// WMMA GEMM: C[nRows,128] (+)= A[nRows,128] @ W^T (+ bias)
// W is pre-converted f16, TRANSPOSED layout [n][k]. Block = 128 thr = 4 waves.
// Each wave computes a 16-row x 128-col strip: 8 col-tiles x 4 K-steps
// = 32 v_wmma_f32_16x16x32_f16 per wave, f32 accumulation.
// ---------------------------------------------------------------------------
__global__ __launch_bounds__(128) void gemm128_wmma(
    const float*    __restrict__ A,
    const _Float16* __restrict__ W,
    const float*    __restrict__ bias,
    float*          __restrict__ C,
    int nRows, int accumulate)
{
    __shared__ __align__(16) _Float16 Wlds[HID * HID];   // [n][k], 32 KB

    const int t = threadIdx.x;

    // ---- stage W into LDS: straight 32KB copy (async path if available) ----
#if USE_ASYNC_LDS
    #pragma unroll
    for (int i = 0; i < 16; ++i) {
        const int idx = (t + i * 128) * 8;               // f16 elems, 16B chunks
        __builtin_amdgcn_global_load_async_to_lds_b128(
            (gas_ptr)(W + idx), (las_ptr)(Wlds + idx), 0, 0);
    }
    __builtin_amdgcn_s_wait_asynccnt(0);
#else
    {
        const float4* __restrict__ src = (const float4*)W;
        float4* dst = (float4*)Wlds;
        #pragma unroll
        for (int i = 0; i < 16; ++i) dst[t + i * 128] = src[t + i * 128];
    }
#endif
    __syncthreads();

    const int wave    = t >> 5;
    const int lane    = t & 31;
    const int rowBase = blockIdx.x * 64 + wave * 16;
    const int m       = lane & 15;    // A-fragment row (M)
    const int kh      = lane >> 4;    // lane half selects K group

    v8f acc[8] = {};

    // branchless: clamp row (garbage rows computed but never stored)
    int arow = rowBase + m;
    if (arow > nRows - 1) arow = nRows - 1;
    const float* __restrict__ Arow = A + (long)arow * HID;

    #pragma unroll
    for (int kt = 0; kt < 4; ++kt) {
        // A fragment 16x32 f16, ISA layout: VGPR v -> K = (v>>2)*16 + kh*8 + (v&3)*2
        // => two contiguous 8-float chunks per lane
        const float4* c0p = (const float4*)(Arow + kt * 32 + kh * 8);
        const float4* c1p = (const float4*)(Arow + kt * 32 + 16 + kh * 8);
        float4 c0a = c0p[0], c0b = c0p[1];
        float4 c1a = c1p[0], c1b = c1p[1];
        v16h afrag;
        afrag[0]  = (_Float16)c0a.x; afrag[1]  = (_Float16)c0a.y;
        afrag[2]  = (_Float16)c0a.z; afrag[3]  = (_Float16)c0a.w;
        afrag[4]  = (_Float16)c0b.x; afrag[5]  = (_Float16)c0b.y;
        afrag[6]  = (_Float16)c0b.z; afrag[7]  = (_Float16)c0b.w;
        afrag[8]  = (_Float16)c1a.x; afrag[9]  = (_Float16)c1a.y;
        afrag[10] = (_Float16)c1a.z; afrag[11] = (_Float16)c1a.w;
        afrag[12] = (_Float16)c1b.x; afrag[13] = (_Float16)c1b.y;
        afrag[14] = (_Float16)c1b.z; afrag[15] = (_Float16)c1b.w;

        // load all 8 B fragments first so one dscnt wait covers 16 ds_load_b128
        v16h bfrag[8];
        #pragma unroll
        for (int ct = 0; ct < 8; ++ct) {
            const int n = ct * 16 + (lane & 15);
            #pragma unroll
            for (int v = 0; v < 8; ++v) {
                const int k = kt * 32 + ((v >> 2) << 4) + (kh << 3) + ((v & 3) << 1);
                bfrag[ct][2 * v]     = Wlds[n * HID + k];
                bfrag[ct][2 * v + 1] = Wlds[n * HID + k + 1];
            }
        }
        #pragma unroll
        for (int ct = 0; ct < 8; ++ct) {
            acc[ct] = __builtin_amdgcn_wmma_f32_16x16x32_f16(
                false, afrag, false, bfrag[ct], (short)0, acc[ct], false, false);
        }
    }

    // C/D layout: VGPR r -> M = (lane/16)*8 + r, N = ct*16 + lane%16
    #pragma unroll
    for (int ct = 0; ct < 8; ++ct) {
        const int col = ct * 16 + (lane & 15);
        #pragma unroll
        for (int r = 0; r < 8; ++r) {
            const int row = rowBase + (lane >> 4) * 8 + r;
            if (row < nRows) {
                float val = acc[ct][r];
                if (bias) val += bias[col];
                float* p = C + (long)row * HID + col;
                if (accumulate) val += *p;
                *p = val;
            }
        }
    }
}

// ---------------------------------------------------------------------------
// Edge scatter: sum[dst] += h[src] (4 floats/thread), cnt[dst] += 1
// ---------------------------------------------------------------------------
__global__ void scatter_add_kernel(const float* __restrict__ h,
                                   const int*   __restrict__ ei,
                                   float* __restrict__ sum,
                                   float* __restrict__ cnt)
{
    long t = (long)blockIdx.x * blockDim.x + threadIdx.x;
    int  e = (int)(t >> 5);
    if (e >= NE) return;
    int c   = (int)(t & 31) << 2;
    int src = ei[e];
    int dst = ei[NE + e];
    const float* hs = h   + (long)src * HID + c;
    float*       sd = sum + (long)dst * HID + c;
    atomicAdd(sd + 0, hs[0]);
    atomicAdd(sd + 1, hs[1]);
    atomicAdd(sd + 2, hs[2]);
    atomicAdd(sd + 3, hs[3]);
    if (c == 0) atomicAdd(cnt + dst, 1.0f);
}

__global__ void mean_kernel(float* __restrict__ sum, const float* __restrict__ cnt) {
    long t = (long)blockIdx.x * blockDim.x + threadIdx.x;
    if (t >= (long)NND * HID) return;
    sum[t] = sum[t] / fmaxf(cnt[t >> 7], 1.0f);
}

// ---------------------------------------------------------------------------
// h = LayerNorm(relu(agg)) — one wave32 per node, shfl_xor reductions
// ---------------------------------------------------------------------------
__global__ __launch_bounds__(128) void relu_ln_kernel(
    const float* __restrict__ agg,
    const float* __restrict__ gamma, const float* __restrict__ beta,
    float* __restrict__ h)
{
    int wave = threadIdx.x >> 5, lane = threadIdx.x & 31;
    int node = blockIdx.x * 4 + wave;
    if (node >= NND) return;
    const float* a = agg + (long)node * HID;
    float v[4];
    float s = 0.0f;
    #pragma unroll
    for (int i = 0; i < 4; ++i) { v[i] = fmaxf(a[lane * 4 + i], 0.0f); s += v[i]; }
    #pragma unroll
    for (int off = 16; off >= 1; off >>= 1) s += __shfl_xor(s, off, 32);
    float mu = s * (1.0f / HID);
    float q = 0.0f;
    #pragma unroll
    for (int i = 0; i < 4; ++i) { float d = v[i] - mu; q += d * d; }
    #pragma unroll
    for (int off = 16; off >= 1; off >>= 1) q += __shfl_xor(q, off, 32);
    float inv = rsqrtf(q * (1.0f / HID) + LN_EPS);
    #pragma unroll
    for (int i = 0; i < 4; ++i) {
        int c = lane * 4 + i;
        h[(long)node * HID + c] = (v[i] - mu) * inv * gamma[c] + beta[c];
    }
}

// ---------------------------------------------------------------------------
// Small dense: out[n,j] = act(in[n,:] @ W[:,j] + b[j])
// ---------------------------------------------------------------------------
__global__ void dense_kernel(const float* __restrict__ in, const float* __restrict__ W,
                             const float* __restrict__ b, float* __restrict__ out,
                             int nRows, int K, int M, int doRelu)
{
    long t = (long)blockIdx.x * blockDim.x + threadIdx.x;
    if (t >= (long)nRows * M) return;
    int n = (int)(t / M), j = (int)(t % M);
    const float* x = in + (long)n * K;
    float acc = b ? b[j] : 0.0f;
    for (int k = 0; k < K; ++k) acc += x[k] * W[k * M + j];
    if (doRelu) acc = fmaxf(acc, 0.0f);
    out[t] = acc;
}

// ---------------------------------------------------------------------------
// global_max_pool via CAS float-max
// ---------------------------------------------------------------------------
__global__ void pool_max_kernel(const float* __restrict__ h,
                                const int* __restrict__ batch,
                                float* __restrict__ g)
{
    long t = (long)blockIdx.x * blockDim.x + threadIdx.x;
    if (t >= (long)NND * HID) return;
    int node = (int)(t >> 7), c = (int)(t & 127);
    float val = h[t];
    float* addr = g + (long)batch[node] * HID + c;
    float old = *addr;
    while (val > old) {
        unsigned int assumed = __float_as_uint(old);
        unsigned int prev = atomicCAS((unsigned int*)addr, assumed, __float_as_uint(val));
        if (prev == assumed) break;
        old = __uint_as_float(prev);
    }
}

// ---------------------------------------------------------------------------
// Host orchestration
// ---------------------------------------------------------------------------
extern "C" void kernel_launch(void* const* d_in, const int* in_sizes, int n_in,
                              void* d_out, int out_size, void* d_ws, size_t ws_size,
                              hipStream_t stream)
{
    // d_in (setup_inputs insertion order, params pytree flattened in order):
    // 0:x 1..3:edge_index0..2 4:batch_vec
    // 5..8 enc(W1,b1,W2,b2)
    // 9..44 layers[l][et]=(W_n,W_r,b)  base = 9 + l*9 + et*3
    // 45..52 norms[l]=(gamma,beta)
    // 53..56 reg_head  57..60 graph_head
    const float* x     = (const float*)d_in[0];
    const int*   ei[3] = {(const int*)d_in[1], (const int*)d_in[2], (const int*)d_in[3]};
    const int*   batch = (const int*)d_in[4];
    auto F = [&](int i) { return (const float*)d_in[i]; };

    // workspace layout
    char* base = (char*)d_ws;
    float* h   = (float*)base;  base += (size_t)NND * HID * sizeof(float);
    float* agg = (float*)base;  base += (size_t)NND * HID * sizeof(float);
    float* scr = (float*)base;  base += (size_t)NND * HID * sizeof(float);  // enc tmp / sum / mean / head tmp
    float* cnt = (float*)base;  base += (size_t)NND * sizeof(float);
    float* g   = (float*)base;  base += (size_t)NG * HID * sizeof(float);
    float* t2  = (float*)base;  base += (size_t)NG * (HID / 2) * sizeof(float);
    _Float16* wbf = (_Float16*)base;      // 25 matrices of 128x128 f16 (transposed)
    auto wmat = [&](int idx) { return wbf + (size_t)idx * HID * HID; };

    const int TPB = 256;
    auto GRID = [](long n) { return (unsigned)((n + 255) / 256); };
    const int gemmGrid = (NND + 63) / 64;

    // --- weight conversion to f16^T (enc W2 + 24 layer matrices) ---
    convert_f16_t_kernel<<<GRID(HID * HID), TPB, 0, stream>>>(F(7), wmat(0), HID * HID);
    for (int l = 0; l < NLAY; ++l)
        for (int et = 0; et < NET; ++et) {
            int pb = 9 + l * 9 + et * 3;
            int wb = 1 + l * 6 + et * 2;
            convert_f16_t_kernel<<<GRID(HID * HID), TPB, 0, stream>>>(F(pb),     wmat(wb),     HID * HID);
            convert_f16_t_kernel<<<GRID(HID * HID), TPB, 0, stream>>>(F(pb + 1), wmat(wb + 1), HID * HID);
        }

    // --- encoder: scr = relu(x@W1+b1) [VALU, K=14]; h = scr@W2+b2 [WMMA] ---
    dense_kernel<<<GRID((long)NND * HID), TPB, 0, stream>>>(x, F(5), F(6), scr, NND, INDIM, HID, 1);
    gemm128_wmma<<<gemmGrid, 128, 0, stream>>>(scr, wmat(0), F(8), h, NND, 0);

    // --- 4 hetero SAGE layers ---
    for (int l = 0; l < NLAY; ++l) {
        for (int et = 0; et < NET; ++et) {
            int pb = 9 + l * 9 + et * 3;
            int wb = 1 + l * 6 + et * 2;
            fill_kernel<<<GRID((long)NND * HID), TPB, 0, stream>>>(scr, 0.0f, (long)NND * HID);
            fill_kernel<<<GRID(NND), TPB, 0, stream>>>(cnt, 0.0f, NND);
            scatter_add_kernel<<<GRID((long)NE * 32), TPB, 0, stream>>>(h, ei[et], scr, cnt);
            mean_kernel<<<GRID((long)NND * HID), TPB, 0, stream>>>(scr, cnt);
            // agg (+)= mean @ W_n + b ; agg += h @ W_r
            gemm128_wmma<<<gemmGrid, 128, 0, stream>>>(scr, wmat(wb), F(pb + 2), agg, NND, et != 0);
            gemm128_wmma<<<gemmGrid, 128, 0, stream>>>(h, wmat(wb + 1), nullptr, agg, NND, 1);
        }
        relu_ln_kernel<<<(NND + 3) / 4, 128, 0, stream>>>(agg, F(45 + l * 2), F(46 + l * 2), h);
    }

    float* node_out  = (float*)d_out;        // [NND]
    float* graph_out = node_out + NND;       // [NG]

    // --- reg head: node_pred = (relu(h@rW1+rb1)) @ rW2 + rb2 ---
    dense_kernel<<<GRID((long)NND * (HID / 2)), TPB, 0, stream>>>(h, F(53), F(54), scr, NND, HID, HID / 2, 1);
    dense_kernel<<<GRID(NND), TPB, 0, stream>>>(scr, F(55), F(56), node_out, NND, HID / 2, 1, 0);

    // --- graph head: g = segment_max(h); graph_pred = mlp2(g) ---
    fill_kernel<<<GRID(NG * HID), TPB, 0, stream>>>(g, -3.402823466e38f, NG * HID);
    pool_max_kernel<<<GRID((long)NND * HID), TPB, 0, stream>>>(h, batch, g);
    dense_kernel<<<GRID(NG * (HID / 2)), TPB, 0, stream>>>(g, F(57), F(58), t2, NG, HID, HID / 2, 1);
    dense_kernel<<<GRID(NG), TPB, 0, stream>>>(t2, F(59), F(60), graph_out, NG, HID / 2, 1, 0);
}
